// Encoder_55293408969294
// MI455X (gfx1250) — matrix operand order repaired and checked by reading, not compiled, not run on hardware
//
#include <hip/hip_runtime.h>

typedef __attribute__((ext_vector_type(16))) _Float16 v16h;
typedef __attribute__((ext_vector_type(8)))  float    v8f;

union Frag {
    v16h v;
    uint4 q[2];
    _Float16 h[16];
};

__device__ __forceinline__ float sigf(float x) { return 1.f / (1.f + __expf(-x)); }
__device__ __forceinline__ float tanhfast(float x) { return 2.f / (1.f + __expf(-2.f * x)) - 1.f; }

// ---------------- Prologue kernels ----------------

// zero h,c ; bias = b_ih + b_hh
__global__ void init_kernel(const float* __restrict__ b_ih, const float* __restrict__ b_hh,
                            float* __restrict__ hc, float* __restrict__ bias) {
    int idx = blockIdx.x * blockDim.x + threadIdx.x;
    if (idx < 262144) hc[idx] = 0.f;          // h (131072) + c (131072)
    if (idx < 1024)   bias[idx] = b_ih[idx] + b_hh[idx];
}

// e_x[b,d] = sum_w x[b,d,w]*w_x[w] + attn_b      (w_x = attn_w[512:768])
__global__ void ex_kernel(const float* __restrict__ x, const float* __restrict__ attn_w,
                          const float* __restrict__ attn_b, float* __restrict__ e_x) {
    int idx = blockIdx.x * blockDim.x + threadIdx.x;   // 0..65535 = b*128+d
    const float4* xr = (const float4*)(x + (size_t)idx * 256);
    const float4* wx = (const float4*)(attn_w + 512);
    float s = 0.f;
    #pragma unroll 8
    for (int j = 0; j < 64; ++j) {
        float4 a = xr[j];
        float4 w = wx[j];
        s += a.x * w.x + a.y * w.y + a.z * w.z + a.w * w.w;
    }
    e_x[idx] = s + attn_b[0];
}

// Pre-swizzle Wcat = [W_ih | W_hh] (1024 x 384) into per-lane WMMA B fragments (f16).
// Fragment (kk, nt, lane): lane<16 -> col N = nt*16+lane, K = kk*32 + 0..15
//                          lane>=16 -> col N = nt*16+lane-16, K = kk*32 + 16..31
__global__ void wswz_kernel(const float* __restrict__ W_ih, const float* __restrict__ W_hh,
                            _Float16* __restrict__ Wsw) {
    int idx = blockIdx.x * blockDim.x + threadIdx.x;   // 0 .. 12*64*32-1
    if (idx >= 12 * 64 * 32) return;
    int lane = idx & 31;
    int nt   = (idx >> 5) & 63;
    int kk   = idx >> 11;
    int N    = nt * 16 + (lane & 15);
    int kb   = kk * 32 + ((lane < 16) ? 0 : 16);
    Frag f;
    #pragma unroll
    for (int e = 0; e < 16; ++e) {
        int K = kb + e;
        float val = (K < 128) ? W_ih[(size_t)N * 128 + K] : W_hh[(size_t)N * 256 + (K - 128)];
        f.h[e] = (_Float16)val;
    }
    uint4* dst = ((uint4*)Wsw) + (size_t)idx * 2;
    dst[0] = f.q[0];
    dst[1] = f.q[1];
}

// x (B*D=65536 rows, W=256 cols) -> xT (256 rows, 65536 cols), LDS tile transpose
__global__ void transpose_kernel(const float* __restrict__ src, float* __restrict__ dst) {
    __shared__ float tile[32][33];
    int x0 = blockIdx.x * 32;   // bd
    int y0 = blockIdx.y * 32;   // t
    int tx = threadIdx.x, ty = threadIdx.y;
    #pragma unroll
    for (int i = 0; i < 32; i += 8)
        tile[ty + i][tx] = src[(size_t)(x0 + ty + i) * 256 + (y0 + tx)];
    __syncthreads();
    #pragma unroll
    for (int i = 0; i < 32; i += 8)
        dst[(size_t)(y0 + ty + i) * 65536 + (x0 + tx)] = tile[tx][ty + i];
}

// ---------------- Per-timestep kernel ----------------
// grid = 16 (32 batch rows each), block = 512 (16 waves)
// wave w owns hcols [w*16, w*16+16) in all 4 gate groups, both 16-row tiles.
__global__ __launch_bounds__(512, 1) void step_kernel(
    const float* __restrict__ x, const float* __restrict__ xT, int use_xt,
    const float* __restrict__ attn_w, const float* __restrict__ e_x,
    const float* __restrict__ bias, const _Float16* __restrict__ Wsw,
    float* __restrict__ h, float* __restrict__ c, float* __restrict__ out, int t) {

    __shared__ __align__(16) _Float16 abuf[32 * 384];   // A = [x_hat | h], 32 rows x K=384
    __shared__ float red[512];
    __shared__ float srow[32], mrow[32], invrow[32];

    const int tid = threadIdx.x;
    const int b0  = blockIdx.x * 32;
    const int r   = tid >> 4;         // 0..31
    const int ch  = tid & 15;
    const int b   = b0 + r;

    // ---- phase 1: s[r] = h[b]·w_h + c[b]·w_c ----
    {
        float p = 0.f;
        const int i0 = ch * 16;
        const float* hr = h + (size_t)b * 256;
        const float* cr = c + (size_t)b * 256;
        #pragma unroll
        for (int i = 0; i < 16; ++i)
            p += hr[i0 + i] * attn_w[i0 + i] + cr[i0 + i] * attn_w[256 + i0 + i];
        red[tid] = p;
    }
    __syncthreads();
    if (ch == 0) {
        float s = 0.f;
        for (int k = 0; k < 16; ++k) s += red[r * 16 + k];
        srow[r] = s;
    }
    __syncthreads();

    // ---- phase 2: softmax over D=128, x_hat -> abuf[:, 0:128] ----
    {
        const float s = srow[r];
        const int d0 = ch * 8;
        const float* exr = e_x + (size_t)b * 128;
        float ev[8];
        float lmax = -3.4e38f;
        #pragma unroll
        for (int j = 0; j < 8; ++j) { ev[j] = exr[d0 + j] + s; lmax = fmaxf(lmax, ev[j]); }
        red[tid] = lmax;
        __syncthreads();
        if (ch == 0) {
            float m = red[r * 16];
            for (int k = 1; k < 16; ++k) m = fmaxf(m, red[r * 16 + k]);
            mrow[r] = m;
        }
        __syncthreads();
        const float m = mrow[r];
        float ls = 0.f;
        #pragma unroll
        for (int j = 0; j < 8; ++j) { ev[j] = __expf(ev[j] - m); ls += ev[j]; }
        __syncthreads();
        red[tid] = ls;
        __syncthreads();
        if (ch == 0) {
            float su = 0.f;
            for (int k = 0; k < 16; ++k) su += red[r * 16 + k];
            invrow[r] = 1.f / su;
        }
        __syncthreads();
        const float inv = invrow[r];
        #pragma unroll
        for (int j = 0; j < 8; ++j) {
            const int d = d0 + j;
            float xv = use_xt ? xT[(size_t)t * 65536 + (size_t)b * 128 + d]
                              : x[((size_t)b * 128 + d) * 256 + t];
            abuf[r * 384 + d] = (_Float16)(ev[j] * inv * xv);
        }
    }
    // ---- h -> abuf[:, 128:384] (f16): 32*256 = 8192 vals, 512 thr x 16 ----
    #pragma unroll
    for (int k = 0; k < 16; ++k) {
        const int e  = tid * 16 + k;           // 0..8191
        const int rr = e >> 8, ii = e & 255;
        abuf[rr * 384 + 128 + ii] = (_Float16)h[(size_t)(b0 + rr) * 256 + ii];
    }
    __syncthreads();

    // ---- phase 3: gates = A @ Wcat^T + bias via WMMA ----
    const int wave = tid >> 5;        // 0..15 -> hcol slab [wave*16, wave*16+16)
    const int lane = tid & 31;
    const int ln15 = lane & 15;

    // prefetch c_old for the epilogue (overlaps with the whole GEMM)
    float cold[16];
    #pragma unroll
    for (int m = 0; m < 2; ++m)
        #pragma unroll
        for (int v = 0; v < 8; ++v) {
            const int rb = m * 16 + v + ((lane >> 4) << 3);
            cold[m * 8 + v] = c[(size_t)(b0 + rb) * 256 + wave * 16 + ln15];
        }

    // acc[(g4<<1)|m] : gate group g4 (i,f,g,o), row tile m
    v8f acc[8];
    #pragma unroll
    for (int g4 = 0; g4 < 4; ++g4) {
        const float bbv = bias[g4 * 256 + wave * 16 + ln15];
        v8f a;
        #pragma unroll
        for (int e = 0; e < 8; ++e) a[e] = bbv;
        acc[(g4 << 1) | 0] = a;
        acc[(g4 << 1) | 1] = a;
    }

    const int koff = (lane < 16) ? 0 : 16;
    const uint4* wsq = (const uint4*)Wsw;

    auto loadB = [&](Frag* dst, int kk) {
        #pragma unroll
        for (int g4 = 0; g4 < 4; ++g4) {
            const int nt = g4 * 16 + wave;
            const uint4* wp = wsq + (size_t)((kk * 64 + nt) * 32 + lane) * 2;
            dst[g4].q[0] = wp[0];
            dst[g4].q[1] = wp[1];
        }
    };

    Frag bb[2][4];
    loadB(bb[0], 0);
    #pragma unroll
    for (int kk = 0; kk < 12; ++kk) {
        Frag* cur = bb[kk & 1];
        Frag* nxt = bb[(kk & 1) ^ 1];
        if (kk + 1 < 12) loadB(nxt, kk + 1);
        Frag af[2];
        #pragma unroll
        for (int m = 0; m < 2; ++m) {
            const int arow = m * 16 + ln15;
            af[m].q[0] = *(const uint4*)&abuf[arow * 384 + kk * 32 + koff];
            af[m].q[1] = *(const uint4*)&abuf[arow * 384 + kk * 32 + koff + 8];
        }
        #pragma unroll
        for (int g4 = 0; g4 < 4; ++g4)
            #pragma unroll
            for (int m = 0; m < 2; ++m)
                acc[(g4 << 1) | m] = __builtin_amdgcn_wmma_f32_16x16x32_f16(
                    false, af[m].v, false, cur[g4].v, (short)0, acc[(g4 << 1) | m],
                    false, false);
    }

    // ---- epilogue: lane-local LSTM cell update ----
    #pragma unroll
    for (int m = 0; m < 2; ++m)
        #pragma unroll
        for (int v = 0; v < 8; ++v) {
            const int rb  = m * 16 + v + ((lane >> 4) << 3);
            const int bbi = b0 + rb;
            const int hcol = wave * 16 + ln15;
            const float iv = acc[(0 << 1) | m][v];
            const float fv = acc[(1 << 1) | m][v];
            const float gv = acc[(2 << 1) | m][v];
            const float ov = acc[(3 << 1) | m][v];
            const size_t hci = (size_t)bbi * 256 + hcol;
            const float cn = sigf(fv) * cold[m * 8 + v] + sigf(iv) * tanhfast(gv);
            const float hn = sigf(ov) * tanhfast(cn);
            c[hci] = cn;
            h[hci] = hn;
            __builtin_nontemporal_store(hn, &out[(size_t)bbi * 65536 + (size_t)t * 256 + hcol]);
        }
}

// ---------------- Host launch ----------------
extern "C" void kernel_launch(void* const* d_in, const int* in_sizes, int n_in,
                              void* d_out, int out_size, void* d_ws, size_t ws_size,
                              hipStream_t stream) {
    const float* x      = (const float*)d_in[0];
    const float* attn_w = (const float*)d_in[1];
    const float* attn_b = (const float*)d_in[2];
    const float* W_ih   = (const float*)d_in[3];
    const float* W_hh   = (const float*)d_in[4];
    const float* b_ih   = (const float*)d_in[5];
    const float* b_hh   = (const float*)d_in[6];
    float* out = (float*)d_out;

    char* ws = (char*)d_ws;
    float*      h    = (float*)(ws + 0);          // 512 KB
    float*      c    = (float*)(ws + 524288);     // 512 KB
    float*      e_x  = (float*)(ws + 1048576);    // 256 KB
    float*      bias = (float*)(ws + 1310720);    // 4 KB
    _Float16*   Wsw  = (_Float16*)(ws + 1314816); // 768 KB
    float*      xT   = (float*)(ws + 2101248);    // optional 64 MB

    const int use_xt = (ws_size >= 2101248ull + 67108864ull) ? 1 : 0;

    init_kernel<<<1024, 256, 0, stream>>>(b_ih, b_hh, h, bias);
    ex_kernel<<<256, 256, 0, stream>>>(x, attn_w, attn_b, e_x);
    wswz_kernel<<<96, 256, 0, stream>>>(W_ih, W_hh, Wsw);
    if (use_xt)
        transpose_kernel<<<dim3(2048, 8), dim3(32, 8), 0, stream>>>(x, xT);

    for (int t = 0; t < 256; ++t)
        step_kernel<<<16, 512, 0, stream>>>(x, xT, use_xt, attn_w, e_x, bias, Wsw,
                                            h, c, out, t);
}
